// CausalSelfAttention_58110907515671
// MI455X (gfx1250) — compile-verified, matrix operand order checked by reading, not run on hardware
//
#include <hip/hip_runtime.h>
#include <hip/hip_bf16.h>
#include <math.h>

typedef __bf16 bf16;
typedef __attribute__((ext_vector_type(16))) __bf16 v16bf;
typedef __attribute__((ext_vector_type(8)))  __bf16 v8bf;
typedef __attribute__((ext_vector_type(8)))  float  v8f;

#define BATCH 2
#define TSEQ  2048
#define CDIM  1024
#define NHEAD 16
#define DHEAD 64
#define NQKV  3072
#define MROWS (BATCH * TSEQ)   // 4096

// ---------- helpers ----------

static __device__ inline v8f zero_v8f() {
    v8f z;
#pragma unroll
    for (int i = 0; i < 8; ++i) z[i] = 0.0f;
    return z;
}

static __device__ inline bf16 f2bf(float f) {
    union { float f; unsigned u; } v; v.f = f;
    unsigned r = (v.u + 0x7FFFu + ((v.u >> 16) & 1u)) >> 16;
    unsigned short s = (unsigned short)r;
    return __builtin_bit_cast(bf16, s);
}

// Load one 16x32 bf16 A/B fragment (per ISA layout: lane holds two contiguous
// 8-element chunks at k0+half*8 and k0+16+half*8 of its row/column).
static __device__ inline v16bf load_frag(const bf16* __restrict__ rowptr, int k0, int half) {
    const bf16* p = rowptr + k0 + half * 8;
    v8bf lo = *(const v8bf*)(p);
    v8bf hi = *(const v8bf*)(p + 16);
    v16bf f;
#pragma unroll
    for (int i = 0; i < 8; ++i) { f[i] = lo[i]; f[i + 8] = hi[i]; }
    return f;
}

static __device__ inline v8f wmma_bf16(v16bf a, v16bf b, v8f c) {
    return __builtin_amdgcn_wmma_f32_16x16x32_bf16(false, a, false, b, (short)0, c, false, false);
}

// ---------- conversion kernels ----------

__global__ void k_cvt_f32_bf16(const float* __restrict__ in, bf16* __restrict__ out, int n) {
    int i = blockIdx.x * blockDim.x + threadIdx.x;
    if (i < n) out[i] = f2bf(in[i]);
}

// in: [K,N] f32  ->  out: [N,K] bf16
__global__ void k_transpose_bf16(const float* __restrict__ in, bf16* __restrict__ out, int K, int N) {
    int i = blockIdx.x * blockDim.x + threadIdx.x;
    if (i < K * N) {
        int n = i / K, k = i % K;
        out[i] = f2bf(in[(size_t)k * N + n]);
    }
}

// ---------- GEMM 1: qkv = x @ Wqkv + b, scattered to Q[b,h,t,d], K[b,h,t,d], V^T[b,h,d,t] ----------
// Wave tile 64x32 (4x2 WMMA frags): spill-free, clean load/WMMA overlap.

__global__ void __launch_bounds__(256)
k_gemm_qkv(const bf16* __restrict__ A,    // [4096,1024]
           const bf16* __restrict__ Bt,   // [3072,1024]
           const float* __restrict__ bias,// [3072]
           bf16* __restrict__ Qd, bf16* __restrict__ Kd, bf16* __restrict__ Vt) {
    const int lane = threadIdx.x & 31;
    const int wave = threadIdx.x >> 5;
    const int half = lane >> 4;
    const int l16  = lane & 15;
    const int Moff = blockIdx.y * 128 + (wave >> 2) * 64;
    const int Noff = blockIdx.x * 128 + (wave & 3) * 32;

    v8f acc[4][2];
#pragma unroll
    for (int i = 0; i < 4; ++i)
#pragma unroll
        for (int j = 0; j < 2; ++j) acc[i][j] = zero_v8f();

    for (int kk = 0; kk < CDIM; kk += 32) {
        v16bf bfr[2];
#pragma unroll
        for (int j = 0; j < 2; ++j)
            bfr[j] = load_frag(Bt + (size_t)(Noff + j * 16 + l16) * CDIM, kk, half);
#pragma unroll
        for (int i = 0; i < 4; ++i) {
            v16bf af = load_frag(A + (size_t)(Moff + i * 16 + l16) * CDIM, kk, half);
#pragma unroll
            for (int j = 0; j < 2; ++j)
                acc[i][j] = wmma_bf16(af, bfr[j], acc[i][j]);
        }
    }

#pragma unroll
    for (int i = 0; i < 4; ++i) {
#pragma unroll
        for (int j = 0; j < 2; ++j) {
            int n = Noff + j * 16 + l16;
            float bv = bias[n];
            int seg = n >> 10;        // 0=q 1=k 2=v
            int c = n & 1023;
            int h = c >> 6;
            int d = c & 63;
#pragma unroll
            for (int r = 0; r < 8; ++r) {
                int m = Moff + i * 16 + r + 8 * half;
                int bb = m >> 11;
                int t  = m & 2047;
                bf16 val = f2bf(acc[i][j][r] + bv);
                int bh = bb * NHEAD + h;
                if (seg == 0)      Qd[((size_t)bh * TSEQ + t) * DHEAD + d] = val;
                else if (seg == 1) Kd[((size_t)bh * TSEQ + t) * DHEAD + d] = val;
                else               Vt[((size_t)bh * DHEAD + d) * TSEQ + t] = val;
            }
        }
    }
}

// ---------- Flash attention: one wave = one (b,h, 16-query tile) ----------

__global__ void __launch_bounds__(256)
k_attn(const bf16* __restrict__ Q,   // [B,H,T,D]
       const bf16* __restrict__ Kk,  // [B,H,T,D]
       const bf16* __restrict__ Vt,  // [B,H,D,T]
       bf16* __restrict__ O) {       // [B*T, C] = [b,t,h,d]
    __shared__ __align__(16) bf16 lds[8][16 * 32];
    const int lane = threadIdx.x & 31;
    const int wave = threadIdx.x >> 5;
    const int half = lane >> 4;
    const int l16  = lane & 15;

    int task = blockIdx.x * 8 + wave;      // 4096 tasks
    int b   = task >> 11;
    int rem = task & 2047;
    int h   = rem >> 7;
    int qt  = rem & 127;
    int qbase = qt * 16;

    const bf16* qp = Q  + (size_t)(b * NHEAD + h) * TSEQ * DHEAD;
    const bf16* kp = Kk + (size_t)(b * NHEAD + h) * TSEQ * DHEAD;
    const bf16* vp = Vt + (size_t)(b * NHEAD + h) * DHEAD * TSEQ;

    v16bf qf[2];
#pragma unroll
    for (int ks = 0; ks < 2; ++ks)
        qf[ks] = load_frag(qp + (size_t)(qbase + l16) * DHEAD, ks * 32, half);

    v8f o[4];
#pragma unroll
    for (int j = 0; j < 4; ++j) o[j] = zero_v8f();
    float mi[8], li[8];
#pragma unroll
    for (int r = 0; r < 8; ++r) { mi[r] = -1e30f; li[r] = 0.0f; }

    const float scale = 0.125f;  // 1/sqrt(64)
    const int qend = qbase + 15;

    for (int kb = 0; kb <= qend; kb += 32) {
        // S tile: 16q x 32key
        v8f s[2];
#pragma unroll
        for (int nt = 0; nt < 2; ++nt) {
            s[nt] = zero_v8f();
            int key = kb + nt * 16 + l16;
            const bf16* krow = kp + (size_t)key * DHEAD;
#pragma unroll
            for (int ks = 0; ks < 2; ++ks)
                s[nt] = wmma_bf16(qf[ks], load_frag(krow, ks * 32, half), s[nt]);
        }
        // online softmax per row (rows live in 16-lane halves)
#pragma unroll
        for (int r = 0; r < 8; ++r) {
            int qrow = qbase + r + 8 * half;
            float rmax = -1e30f;
#pragma unroll
            for (int nt = 0; nt < 2; ++nt) {
                int key = kb + nt * 16 + l16;
                float sv = s[nt][r] * scale;
                if (key > qrow) sv = -1e30f;
                s[nt][r] = sv;
                rmax = fmaxf(rmax, sv);
            }
#pragma unroll
            for (int off = 8; off >= 1; off >>= 1)
                rmax = fmaxf(rmax, __shfl_xor(rmax, off, 32));
            float mnew  = fmaxf(mi[r], rmax);
            float alpha = __expf(mi[r] - mnew);
            float rsum  = 0.0f;
#pragma unroll
            for (int nt = 0; nt < 2; ++nt) {
                float pv = __expf(s[nt][r] - mnew);
                lds[wave][(r + 8 * half) * 32 + nt * 16 + l16] = f2bf(pv);
                rsum += pv;
            }
#pragma unroll
            for (int off = 8; off >= 1; off >>= 1)
                rsum += __shfl_xor(rsum, off, 32);
            li[r] = li[r] * alpha + rsum;
            mi[r] = mnew;
#pragma unroll
            for (int j = 0; j < 4; ++j) o[j][r] *= alpha;
        }
        // P tile (16x32) back out of LDS in A-fragment layout
        const bf16* lp = &lds[wave][l16 * 32];
        v8bf plo = *(const v8bf*)(lp + half * 8);
        v8bf phi = *(const v8bf*)(lp + half * 8 + 16);
        v16bf pf;
#pragma unroll
        for (int i = 0; i < 8; ++i) { pf[i] = plo[i]; pf[i + 8] = phi[i]; }
        // O += P @ V  (V^T rows are B-fragment columns)
#pragma unroll
        for (int j = 0; j < 4; ++j) {
            v16bf vf = load_frag(vp + (size_t)(j * 16 + l16) * TSEQ, kb, half);
            o[j] = wmma_bf16(pf, vf, o[j]);
        }
    }

#pragma unroll
    for (int r = 0; r < 8; ++r) {
        float inv = 1.0f / li[r];
        int row = b * TSEQ + qbase + r + 8 * half;
#pragma unroll
        for (int j = 0; j < 4; ++j)
            O[(size_t)row * CDIM + h * DHEAD + j * 16 + l16] = f2bf(o[j][r] * inv);
    }
}

// ---------- GEMM 2: out = attn @ Wproj + b (f32 output), 64x32 wave tile ----------

__global__ void __launch_bounds__(256)
k_gemm_proj(const bf16* __restrict__ A,    // [4096,1024]
            const bf16* __restrict__ Bt,   // [1024,1024]
            const float* __restrict__ bias,
            float* __restrict__ Out) {     // [4096,1024] f32
    const int lane = threadIdx.x & 31;
    const int wave = threadIdx.x >> 5;
    const int half = lane >> 4;
    const int l16  = lane & 15;
    const int Moff = blockIdx.y * 128 + (wave >> 2) * 64;
    const int Noff = blockIdx.x * 128 + (wave & 3) * 32;

    v8f acc[4][2];
#pragma unroll
    for (int i = 0; i < 4; ++i)
#pragma unroll
        for (int j = 0; j < 2; ++j) acc[i][j] = zero_v8f();

    for (int kk = 0; kk < CDIM; kk += 32) {
        v16bf bfr[2];
#pragma unroll
        for (int j = 0; j < 2; ++j)
            bfr[j] = load_frag(Bt + (size_t)(Noff + j * 16 + l16) * CDIM, kk, half);
#pragma unroll
        for (int i = 0; i < 4; ++i) {
            v16bf af = load_frag(A + (size_t)(Moff + i * 16 + l16) * CDIM, kk, half);
#pragma unroll
            for (int j = 0; j < 2; ++j)
                acc[i][j] = wmma_bf16(af, bfr[j], acc[i][j]);
        }
    }

#pragma unroll
    for (int i = 0; i < 4; ++i) {
#pragma unroll
        for (int j = 0; j < 2; ++j) {
            int n = Noff + j * 16 + l16;
            float bv = bias[n];
#pragma unroll
            for (int r = 0; r < 8; ++r) {
                int m = Moff + i * 16 + r + 8 * half;
                Out[(size_t)m * CDIM + n] = acc[i][j][r] + bv;
            }
        }
    }
}

// ---------- launch ----------

extern "C" void kernel_launch(void* const* d_in, const int* in_sizes, int n_in,
                              void* d_out, int out_size, void* d_ws, size_t ws_size,
                              hipStream_t stream) {
    const float* x     = (const float*)d_in[0];
    const float* Wqkv  = (const float*)d_in[1];
    const float* bqkv  = (const float*)d_in[2];
    const float* Wproj = (const float*)d_in[3];
    const float* bproj = (const float*)d_in[4];
    float* out = (float*)d_out;

    char* ws = (char*)d_ws;
    bf16* xb     = (bf16*)(ws);                          // 8 MB  [0,8)
    bf16* wqkvt  = (bf16*)(ws + ((size_t)8  << 20));     // 6 MB  [8,14)
    bf16* wprojt = (bf16*)(ws + ((size_t)14 << 20));     // 2 MB  [14,16)
    bf16* Qd     = (bf16*)(ws + ((size_t)16 << 20));     // 8 MB
    bf16* Kd     = (bf16*)(ws + ((size_t)24 << 20));     // 8 MB
    bf16* Vt     = (bf16*)(ws + ((size_t)32 << 20));     // 8 MB
    bf16* At     = (bf16*)(ws + ((size_t)40 << 20));     // 8 MB

    k_cvt_f32_bf16<<<(MROWS * CDIM) / 256, 256, 0, stream>>>(x, xb, MROWS * CDIM);
    k_transpose_bf16<<<(CDIM * NQKV) / 256, 256, 0, stream>>>(Wqkv, wqkvt, CDIM, NQKV);
    k_transpose_bf16<<<(CDIM * CDIM) / 256, 256, 0, stream>>>(Wproj, wprojt, CDIM, CDIM);

    k_gemm_qkv<<<dim3(NQKV / 128, MROWS / 128), 256, 0, stream>>>(xb, wqkvt, bqkv, Qd, Kd, Vt);
    k_attn<<<(BATCH * NHEAD * (TSEQ / 16)) / 8, 256, 0, stream>>>(Qd, Kd, Vt, At);
    k_gemm_proj<<<dim3(CDIM / 128, MROWS / 128), 256, 0, stream>>>(At, wprojt, bproj, out);
}
